// BiasedAxialAttention_38457137168472
// MI455X (gfx1250) — matrix operand order repaired and checked by reading, not compiled, not run on hardware
//
#include <hip/hip_runtime.h>

// ---------------------------------------------------------------------------
// BiasedAxialAttention (triangle attention, row-wise) for MI455X / gfx1250.
// B=1, L=256, DP=DB=128, H=8, DH=32.
//
// All contractions use v_wmma_f32_16x16x32_bf16 (wave32).  Operand buffers
// are laid out so WMMA fragments are CONTIGUOUS per lane:
//   A fragment (16x32 bf16): two 8-element runs  -> 2x b128 loads
//   B fragment (32x16 bf16): one 16-element run  -> 1x 32B load (2x b128)
// The K=8192 logits GEMM additionally stages operand tiles in LDS with
// double-buffered gfx1250 async loads (global_load_async_to_lds_*,
// s_wait_asynccnt) to cut L2 traffic ~2.6x and overlap copy with WMMA.
// ---------------------------------------------------------------------------

typedef __attribute__((ext_vector_type(16))) __bf16 v16bf;
typedef __attribute__((ext_vector_type(8)))  __bf16 v8bf;
typedef __attribute__((ext_vector_type(4)))  __bf16 v4bf;
typedef __attribute__((ext_vector_type(8)))  float  v8f;

static __device__ __forceinline__ float nan0(float x) {  // jnp.nan_to_num
  if (x != x) return 0.f;
  if (x >  3.402823466e38f) return  3.402823466e38f;
  if (x < -3.402823466e38f) return -3.402823466e38f;
  return x;
}

// A fragment (16-bit A 16x32, this file's contiguous k order, same as B): lane holds row m = lane&15;
// element e -> k = k0 + half*16 + e  == two contiguous runs.
static __device__ __forceinline__ v16bf load_a_frag(const __bf16* __restrict__ row,
                                                    int k0, int half) {
  const v8bf lo = *(const v8bf*)(row + k0 + half * 16);
  const v8bf hi = *(const v8bf*)(row + k0 + half * 16 + 8);
  return __builtin_shufflevector(lo, hi, 0, 1, 2, 3, 4, 5, 6, 7,
                                 8, 9, 10, 11, 12, 13, 14, 15);
}
// B fragment (32x16): lane holds col n = lane&15; element e -> k = k0+half*16+e.
static __device__ __forceinline__ v16bf load_b_frag(const __bf16* __restrict__ col,
                                                    int k0, int half) {
  return *(const v16bf*)(col + k0 + half * 16);
}

// gfx1250 async global->LDS copies (ASYNCcnt-tracked).  VDST = LDS byte
// offset (flat shared address truncated to 32 bits, ISA 10.2), VADDR = 64-bit
// global address.
static __device__ __forceinline__ void async_b128(__bf16* lds_dst,
                                                  const __bf16* gsrc) {
  const unsigned loff = (unsigned)(size_t)lds_dst;
  asm volatile("global_load_async_to_lds_b128 %0, %1, off"
               :: "v"(loff), "v"(gsrc) : "memory");
}
static __device__ __forceinline__ void async_b64(__bf16* lds_dst,
                                                 const __bf16* gsrc) {
  const unsigned loff = (unsigned)(size_t)lds_dst;
  asm volatile("global_load_async_to_lds_b64 %0, %1, off"
               :: "v"(loff), "v"(gsrc) : "memory");
}

// ---------------------------------------------------------------------------
// K1: LayerNorm(pair) with transpose fused.  pn[n][i][c] = LN(pair[i][n][:]).
// ---------------------------------------------------------------------------
__global__ void ln_pair_kernel(const float* __restrict__ pair,
                               const float* __restrict__ g,
                               const float* __restrict__ b,
                               __bf16* __restrict__ pn) {
  const int lane = threadIdx.x & 31;
  const int wave = threadIdx.x >> 5;
  const int r = blockIdx.x * 8 + wave;      // r = n*256 + i
  const int n = r >> 8, i2 = r & 255;
  const float4* src = (const float4*)(pair + (size_t)(i2 * 256 + n) * 128);
  float4 x4 = src[lane];
  float x[4] = { nan0(x4.x), nan0(x4.y), nan0(x4.z), nan0(x4.w) };
  float s  = x[0] + x[1] + x[2] + x[3];
  float sq = x[0]*x[0] + x[1]*x[1] + x[2]*x[2] + x[3]*x[3];
  #pragma unroll
  for (int m = 16; m >= 1; m >>= 1) {
    s  += __shfl_xor(s,  m, 32);
    sq += __shfl_xor(sq, m, 32);
  }
  const float mean = s * (1.f / 128.f);
  const float var  = sq * (1.f / 128.f) - mean * mean;
  const float rstd = rsqrtf(var + 1e-5f);
  const int c = lane * 4;
  v4bf o;
  o[0] = (__bf16)((x[0] - mean) * rstd * g[c + 0] + b[c + 0]);
  o[1] = (__bf16)((x[1] - mean) * rstd * g[c + 1] + b[c + 1]);
  o[2] = (__bf16)((x[2] - mean) * rstd * g[c + 2] + b[c + 2]);
  o[3] = (__bf16)((x[3] - mean) * rstd * g[c + 3] + b[c + 3]);
  *(v4bf*)(pn + (size_t)r * 128 + c) = o;
}

// ---------------------------------------------------------------------------
// K2: LayerNorm(bias) fused with bn @ Wb -> b_proj[r][8]  (bn never stored).
// ---------------------------------------------------------------------------
__global__ void ln_bias_bproj_kernel(const float* __restrict__ bias,
                                     const float* __restrict__ g,
                                     const float* __restrict__ b,
                                     const float* __restrict__ Wb,
                                     float* __restrict__ bproj) {
  const int lane = threadIdx.x & 31;
  const int wave = threadIdx.x >> 5;
  const int r = blockIdx.x * 8 + wave;
  const int n = r >> 8, i2 = r & 255;
  const float4* src = (const float4*)(bias + (size_t)(i2 * 256 + n) * 128);
  float4 x4 = src[lane];
  float x[4] = { nan0(x4.x), nan0(x4.y), nan0(x4.z), nan0(x4.w) };
  float s  = x[0] + x[1] + x[2] + x[3];
  float sq = x[0]*x[0] + x[1]*x[1] + x[2]*x[2] + x[3]*x[3];
  #pragma unroll
  for (int m = 16; m >= 1; m >>= 1) {
    s  += __shfl_xor(s,  m, 32);
    sq += __shfl_xor(sq, m, 32);
  }
  const float mean = s * (1.f / 128.f);
  const float var  = sq * (1.f / 128.f) - mean * mean;
  const float rstd = rsqrtf(var + 1e-5f);
  const int c = lane * 4;
  float ln[4];
  #pragma unroll
  for (int k = 0; k < 4; ++k) ln[k] = (x[k] - mean) * rstd * g[c + k] + b[c + k];
  float ph[8];
  #pragma unroll
  for (int h = 0; h < 8; ++h) {
    float p = 0.f;
    #pragma unroll
    for (int k = 0; k < 4; ++k) p += ln[k] * Wb[(size_t)(c + k) * 8 + h];
    #pragma unroll
    for (int m = 16; m >= 1; m >>= 1) p += __shfl_xor(p, m, 32);
    ph[h] = p;
  }
  if (lane == 0) {
    #pragma unroll
    for (int h = 0; h < 8; ++h) bproj[(size_t)r * 8 + h] = ph[h];
  }
}

// ---------------------------------------------------------------------------
// K3: weights -> bf16, TRANSPOSED (K-contiguous per output column).
// ---------------------------------------------------------------------------
__global__ void convert_weights_kernel(const float* __restrict__ Wq,
                                       const float* __restrict__ Wk,
                                       const float* __restrict__ Wv,
                                       const float* __restrict__ Wg,
                                       const float* __restrict__ Wo,
                                       __bf16* __restrict__ Wall_t,
                                       __bf16* __restrict__ WoT) {
  const int idx = blockIdx.x * 256 + threadIdx.x;     // 0 .. 163839
  if (idx < 131072) {
    const int c = idx >> 7, k = idx & 127;            // Wall_t[c][k]
    const float* src = (c < 256) ? Wq : (c < 512) ? Wk : (c < 768) ? Wv : Wg;
    Wall_t[idx] = (__bf16)src[(size_t)k * 256 + (c & 255)];
  } else {
    const int j = idx - 131072;                       // WoT[c][kk]
    const int c = j >> 8, kk = j & 255;
    WoT[j] = (__bf16)Wo[(size_t)kk * 128 + c];
  }
}

// ---------------------------------------------------------------------------
// K4: projection GEMM  (65536 x 1024, K=128): pn @ [Wq|Wk|Wv|Wg].
// Epilogue scatters into WMMA-friendly layouts.
// ---------------------------------------------------------------------------
__global__ void proj_gemm_kernel(const __bf16* __restrict__ pn,
                                 const __bf16* __restrict__ Wall_t,
                                 const float* __restrict__ bg,
                                 __bf16* __restrict__ q_ws,
                                 __bf16* __restrict__ k_ws,
                                 __bf16* __restrict__ v_ws,
                                 __bf16* __restrict__ gate_ws) {
  const int lane = threadIdx.x & 31;
  const int wid  = blockIdx.x * 8 + (threadIdx.x >> 5);   // 0..262143
  const int mt = wid >> 6, nt = wid & 63;
  const int half = lane >> 4, lr = lane & 15;
  const int m = mt * 16 + lr;
  const int n = nt * 16 + lr;
  const __bf16* arow = pn + (size_t)m * 128;
  const __bf16* bcol = Wall_t + (size_t)n * 128;
  v8f acc = {};
  #pragma unroll
  for (int k0 = 0; k0 < 128; k0 += 32) {
    v16bf a = load_a_frag(arow, k0, half);
    v16bf bb = load_b_frag(bcol, k0, half);
    acc = __builtin_amdgcn_wmma_f32_16x16x32_bf16(false, a, false, bb,
                                                  (short)0, acc, false, false);
  }
  const int cls = n >> 8;            // 0=q 1=k 2=v 3=gate (uniform per lane)
  const int h = (n >> 5) & 7;
  const int d = n & 31;
  const float bgv = (cls == 3) ? bg[n - 768] : 0.f;
  #pragma unroll
  for (int rr = 0; rr < 8; ++rr) {
    const int r = mt * 16 + rr + half * 8;   // r = nn*256 + i2
    const int i2 = r & 255, nn = r >> 8;
    float v = acc[rr];
    if (cls == 3) v = 1.f / (1.f + expf(-(v + bgv)));
    const __bf16 bv = (__bf16)v;
    if (cls == 0)      q_ws[(size_t)(h * 256 + i2) * 8192 + nn * 32 + d] = bv;
    else if (cls == 1) k_ws[(size_t)(h * 256 + i2) * 8192 + nn * 32 + d] = bv;
    else if (cls == 2) v_ws[((size_t)(h * 256 + nn) * 32 + d) * 256 + i2] = bv;
    else               gate_ws[(size_t)r * 256 + h * 32 + d] = bv;
  }
}

// ---------------------------------------------------------------------------
// K5: attention logits.  Head h: C[i,j] = sum_{n,d} q*k, K = 8192.
// Block = 256 thr computes 32(i) x 64(j) for one head; A/B tiles staged in
// LDS with double-buffered async global->LDS copies.  Waves: 2x4 grid of
// 16x16 WMMA tiles.  Every thread issues exactly 2 async ops per stage, so
// s_wait_asynccnt 2 after issuing stage N+1 guarantees stage N is resident.
// ---------------------------------------------------------------------------
__global__ void attn_logits_kernel(const __bf16* __restrict__ q_ws,
                                   const __bf16* __restrict__ k_ws,
                                   const float* __restrict__ bproj,
                                   float* __restrict__ logits) {
  const float SCALE = 0.011048543456039805f;  // 1/(sqrt(32)*sqrt(256))
  __shared__ __attribute__((aligned(128))) __bf16 As[2][32 * 32];  // 2x2KB
  __shared__ __attribute__((aligned(128))) __bf16 Bs[2][64 * 32];  // 2x4KB

  const int t    = threadIdx.x;
  const int lane = t & 31;
  const int wave = t >> 5;
  const int bid  = blockIdx.x;                 // 0..255
  const int h  = bid >> 5;                     // 8 heads
  const int i0 = ((bid >> 2) & 7) * 32;        // 8 i-blocks of 32
  const int j0 = (bid & 3) * 64;               // 4 j-blocks of 64
  const int wi = wave >> 2, wj = wave & 3;     // 2x4 wave grid
  const int half = lane >> 4, lr = lane & 15;

  const __bf16* qbase = q_ws + (size_t)(h * 256) * 8192;
  const __bf16* kbase = k_ws + (size_t)(h * 256) * 8192;

  // staging roles (all 256 threads, 2 async ops each per stage)
  const int brow = t >> 2, bseg = t & 3;       // B: 64 cols x 16B
  const int arow = t >> 3, aseg = t & 7;       // A: 32 rows x 8B

  v8f acc = {};
  // prologue: stage k0 = 0 into buffer 0
  async_b128(&Bs[0][brow * 32 + bseg * 8],
             kbase + (size_t)(j0 + brow) * 8192 + bseg * 8);
  async_b64(&As[0][arow * 32 + aseg * 4],
            qbase + (size_t)(i0 + arow) * 8192 + aseg * 4);

  int buf = 0;
  for (int k0 = 0; k0 < 8192; k0 += 32) {
    if (k0 + 32 < 8192) {
      const int kn = k0 + 32;
      async_b128(&Bs[buf ^ 1][brow * 32 + bseg * 8],
                 kbase + (size_t)(j0 + brow) * 8192 + kn + bseg * 8);
      async_b64(&As[buf ^ 1][arow * 32 + aseg * 4],
                qbase + (size_t)(i0 + arow) * 8192 + kn + aseg * 4);
      asm volatile("s_wait_asynccnt 2" ::: "memory");
    } else {
      asm volatile("s_wait_asynccnt 0" ::: "memory");
    }
    __syncthreads();
    {
      const __bf16* ar = &As[buf][(wi * 16 + lr) * 32];
      const v8bf lo = *(const v8bf*)(ar + half * 16);
      const v8bf hi = *(const v8bf*)(ar + half * 16 + 8);
      v16bf a = __builtin_shufflevector(lo, hi, 0, 1, 2, 3, 4, 5, 6, 7,
                                        8, 9, 10, 11, 12, 13, 14, 15);
      v16bf bb = *(const v16bf*)(&Bs[buf][(wj * 16 + lr) * 32 + half * 16]);
      acc = __builtin_amdgcn_wmma_f32_16x16x32_bf16(false, a, false, bb,
                                                    (short)0, acc, false, false);
    }
    __syncthreads();
    buf ^= 1;
  }
  #pragma unroll
  for (int rr = 0; rr < 8; ++rr) {
    const int i = i0 + wi * 16 + rr + half * 8;
    const int j = j0 + wj * 16 + lr;
    logits[(size_t)(h * 256 + i) * 256 + j] =
        acc[rr] * SCALE + bproj[(size_t)(i * 256 + j) * 8 + h];
  }
}

// ---------------------------------------------------------------------------
// K6: masked softmax over j for each (h,i).  One wave per row.
// ---------------------------------------------------------------------------
__global__ void softmax_kernel(const float* __restrict__ logits,
                               const int* __restrict__ mask,
                               __bf16* __restrict__ attn) {
  const int lane = threadIdx.x & 31;
  const int row  = blockIdx.x * 8 + (threadIdx.x >> 5);   // 0..2047
  const int i = row & 255;
  const bool mi = (mask[i] == 0);
  float v[8];
  #pragma unroll
  for (int t = 0; t < 8; ++t) {
    const int j = t * 32 + lane;
    const bool mj = (mask[j] == 0);
    v[t] = (mi || mj) ? -1e9f : logits[(size_t)row * 256 + j];
  }
  float mx = v[0];
  #pragma unroll
  for (int t = 1; t < 8; ++t) mx = fmaxf(mx, v[t]);
  #pragma unroll
  for (int m = 16; m >= 1; m >>= 1) mx = fmaxf(mx, __shfl_xor(mx, m, 32));
  float s = 0.f;
  #pragma unroll
  for (int t = 0; t < 8; ++t) { v[t] = expf(v[t] - mx); s += v[t]; }
  #pragma unroll
  for (int m = 16; m >= 1; m >>= 1) s += __shfl_xor(s, m, 32);
  const float inv = 1.f / s;
  #pragma unroll
  for (int t = 0; t < 8; ++t)
    attn[(size_t)row * 256 + t * 32 + lane] = (__bf16)(v[t] * inv);
}

// ---------------------------------------------------------------------------
// K7: out[i,kp,h,d] = sum_j attn_h[i,j] * v[kp,j,h,d], gated.  K=256.
// ---------------------------------------------------------------------------
__global__ void out_gemm_kernel(const __bf16* __restrict__ attn,
                                const __bf16* __restrict__ v_ws,
                                const __bf16* __restrict__ gate_ws,
                                __bf16* __restrict__ go) {
  const int lane = threadIdx.x & 31;
  const int wid  = blockIdx.x * 8 + (threadIdx.x >> 5);   // 0..65535
  const int dt = wid & 1, it = (wid >> 1) & 15, kp = (wid >> 5) & 255, h = wid >> 13;
  const int half = lane >> 4, lr = lane & 15;
  const __bf16* arow = attn + (size_t)(h * 256 + it * 16 + lr) * 256;
  const __bf16* bcol = v_ws + ((size_t)(h * 256 + kp) * 32 + dt * 16 + lr) * 256;
  v8f acc = {};
  #pragma unroll
  for (int k0 = 0; k0 < 256; k0 += 32) {
    v16bf a = load_a_frag(arow, k0, half);
    v16bf bb = load_b_frag(bcol, k0, half);
    acc = __builtin_amdgcn_wmma_f32_16x16x32_bf16(false, a, false, bb,
                                                  (short)0, acc, false, false);
  }
  #pragma unroll
  for (int rr = 0; rr < 8; ++rr) {
    const int i = it * 16 + rr + half * 8;
    const int d = dt * 16 + lr;
    const float gate = (float)gate_ws[(size_t)(i * 256 + kp) * 256 + h * 32 + d];
    go[(size_t)(i * 256 + kp) * 256 + h * 32 + d] = (__bf16)(acc[rr] * gate);
  }
}

// ---------------------------------------------------------------------------
// K8: final GEMM (65536 x 128, K=256) + bo, transposed store with mask zero.
// ---------------------------------------------------------------------------
__global__ void final_gemm_kernel(const __bf16* __restrict__ go,
                                  const __bf16* __restrict__ WoT,
                                  const float* __restrict__ bo,
                                  const int* __restrict__ mask,
                                  float* __restrict__ out) {
  const int lane = threadIdx.x & 31;
  const int wid  = blockIdx.x * 8 + (threadIdx.x >> 5);   // 0..32767
  const int mt = wid >> 3, nt = wid & 7;
  const int half = lane >> 4, lr = lane & 15;
  const __bf16* arow = go + (size_t)(mt * 16 + lr) * 256;
  const __bf16* bcol = WoT + (size_t)(nt * 16 + lr) * 256;
  v8f acc = {};
  #pragma unroll
  for (int k0 = 0; k0 < 256; k0 += 32) {
    v16bf a = load_a_frag(arow, k0, half);
    v16bf bb = load_b_frag(bcol, k0, half);
    acc = __builtin_amdgcn_wmma_f32_16x16x32_bf16(false, a, false, bb,
                                                  (short)0, acc, false, false);
  }
  #pragma unroll
  for (int rr = 0; rr < 8; ++rr) {
    const int rg = mt * 16 + rr + half * 8;      // r = i*256 + kp
    const int i = rg >> 8, kp = rg & 255;
    const int c = nt * 16 + lr;
    float v = acc[rr] + bo[c];
    if (mask[i] == 0 || mask[kp] == 0) v = 0.f;
    out[(size_t)(kp * 256 + i) * 128 + c] = v;
  }
}

// ---------------------------------------------------------------------------
// Launcher.  Workspace layout (bytes), total ~182 MiB:
//   pn      @ 0          16777216   bf16 [65536][128]
//   q_ws    @ 16777216   33554432   bf16 [8][256][8192]   ([h][i][n*32+d])
//   k_ws    @ 50331648   33554432   bf16 [8][256][8192]   ([h][j][n*32+d])
//   v_ws    @ 83886080   33554432   bf16 [8][256][32][256]([h][kp][d][j])
//   gate_ws @ 117440512  33554432   bf16 [65536][256]
//   go      @ 150994944  33554432   bf16 [65536][256]
//   bproj   @ 184549376  2097152    f32  [65536][8]
//   logits  @ 186646528  2097152    f32  [8][256][256]
//   attn    @ 188743680  1048576    bf16 [8][256][256]
//   Wall_t  @ 189792256  262144     bf16 [1024][128]
//   WoT     @ 190054400  65536      bf16 [128][256]
// ---------------------------------------------------------------------------
extern "C" void kernel_launch(void* const* d_in, const int* in_sizes, int n_in,
                              void* d_out, int out_size, void* d_ws, size_t ws_size,
                              hipStream_t stream) {
  const float* pair   = (const float*)d_in[0];
  const float* bias   = (const float*)d_in[1];
  const int*   mask   = (const int*)  d_in[2];
  const float* g_pair = (const float*)d_in[3];
  const float* b_pair = (const float*)d_in[4];
  const float* g_bias = (const float*)d_in[5];
  const float* b_bias = (const float*)d_in[6];
  const float* Wq     = (const float*)d_in[7];
  const float* Wk     = (const float*)d_in[8];
  const float* Wv     = (const float*)d_in[9];
  const float* Wb     = (const float*)d_in[10];
  const float* Wg     = (const float*)d_in[11];
  const float* bg     = (const float*)d_in[12];
  const float* Wo     = (const float*)d_in[13];
  const float* bo     = (const float*)d_in[14];
  float* out = (float*)d_out;

  char* ws = (char*)d_ws;
  __bf16* pn      = (__bf16*)(ws + 0);
  __bf16* q_ws    = (__bf16*)(ws + 16777216);
  __bf16* k_ws    = (__bf16*)(ws + 50331648);
  __bf16* v_ws    = (__bf16*)(ws + 83886080);
  __bf16* gate_ws = (__bf16*)(ws + 117440512);
  __bf16* go      = (__bf16*)(ws + 150994944);
  float*  bproj   = (float*) (ws + 184549376);
  float*  logits  = (float*) (ws + 186646528);
  __bf16* attn    = (__bf16*)(ws + 188743680);
  __bf16* Wall_t  = (__bf16*)(ws + 189792256);
  __bf16* WoT     = (__bf16*)(ws + 190054400);

  ln_pair_kernel      <<<8192,  256, 0, stream>>>(pair, g_pair, b_pair, pn);
  ln_bias_bproj_kernel<<<8192,  256, 0, stream>>>(bias, g_bias, b_bias, Wb, bproj);
  convert_weights_kernel<<<640, 256, 0, stream>>>(Wq, Wk, Wv, Wg, Wo, Wall_t, WoT);
  proj_gemm_kernel    <<<32768, 256, 0, stream>>>(pn, Wall_t, bg,
                                                  q_ws, k_ws, v_ws, gate_ws);
  attn_logits_kernel  <<<256,   256, 0, stream>>>(q_ws, k_ws, bproj, logits);
  softmax_kernel      <<<256,   256, 0, stream>>>(logits, mask, attn);
  out_gemm_kernel     <<<8192,  256, 0, stream>>>(attn, v_ws, gate_ws, go);
  final_gemm_kernel   <<<4096,  256, 0, stream>>>(go, WoT, bo, mask, out);
}
